// AttentionLayer2D_84224308675266
// MI455X (gfx1250) — matrix operand order repaired and checked by reading, not compiled, not run on hardware
//
#include <hip/hip_runtime.h>
#include <math.h>

// Problem constants (reference: B=4, H=W=64, C=512, CFG=64)
#define BN   4
#define NN   4096
#define CC   512
#define CFG_ 64

typedef __attribute__((ext_vector_type(16))) __bf16 v16bf;
typedef __attribute__((ext_vector_type(8)))  __bf16 v8bf;
typedef __attribute__((ext_vector_type(8)))  float  v8f;

// Pack two contiguous 8-element bf16 runs into the 16-bit WMMA operand layout.
__device__ __forceinline__ v16bf pack2(const __bf16* p0, const __bf16* p1) {
    v8bf a = *(const v8bf*)p0;
    v8bf b = *(const v8bf*)p1;
    v16bf r;
#pragma unroll
    for (int j = 0; j < 8; ++j) { r[j] = a[j]; r[j + 8] = b[j]; }
    return r;
}

// ---------------------------------------------------------------------------
// Weight pack: W f32 [Cin=512, Co] -> bf16 in B-operand order:
//   wp[((nt*(Cin/32) + kc)*32 + lane)*16 + j] = W[kc*32 + (lane>>4)*16 + j][nt*16 + lane%16]
// so a GEMM wave reads one contiguous 32B v16bf per (tile, k-step, lane).
// ---------------------------------------------------------------------------
__global__ __launch_bounds__(256)
void pack_w(const float* __restrict__ w, __bf16* __restrict__ wp, int Co) {
    const int i = blockIdx.x * 256 + threadIdx.x;       // total = Cin*Co
    if (i >= CC * Co) return;
    const int j    = i & 15;
    const int lane = (i >> 4) & 31;
    const int g    = i >> 9;                            // (nt, kc) combined
    const int kc   = g % (CC / 32);
    const int nt   = g / (CC / 32);
    const int k = kc * 32 + (lane >> 4) * 16 + j;
    const int n = nt * 16 + (lane & 15);
    wp[i] = (__bf16)w[(size_t)k * Co + n];
}

// ---------------------------------------------------------------------------
// Projection GEMM: Y[N,Co] = X[N,512] @ W + bias, bf16 out.
// One wave: 16 rows x (16*NTW) cols, A operand shared across NTW WMMAs/k-step.
// TRANS=false -> row-major [B,N,Co];  TRANS=true -> [B,Co,N] (v8bf store).
// ---------------------------------------------------------------------------
template<int NTW, bool TRANS>
__global__ __launch_bounds__(256)
void proj_kernel(const float* __restrict__ x, const __bf16* __restrict__ wp,
                 const float* __restrict__ bias, __bf16* __restrict__ out, int Co) {
    const int lane = threadIdx.x & 31, wid = threadIdx.x >> 5;
    const int lq = lane & 15, lh = lane >> 4;
    const int groupsN  = Co / (16 * NTW);
    const int perBatch = (NN >> 4) * groupsN;
    const int gw = blockIdx.x * (blockDim.x >> 5) + wid;
    const int b  = gw / perBatch;
    if (b >= BN) return;                     // wave-uniform, EXEC stays all-1
    const int t  = gw % perBatch;
    const int mt = t / groupsN, ng = t % groupsN;

    const float* xb = x + (size_t)b * NN * CC;
    const int arow = mt * 16 + lq;

    v8f acc[NTW];
#pragma unroll
    for (int u = 0; u < NTW; ++u) acc[u] = (v8f){};

    for (int kc = 0; kc < CC; kc += 32) {
        // A: x rows f32 -> bf16, elem j -> k = kc + lh*16 + j
        v8f x0 = *(const v8f*)(xb + (size_t)arow * CC + kc + lh * 16);
        v8f x1 = *(const v8f*)(xb + (size_t)arow * CC + kc + lh * 16 + 8);
        v16bf a;
#pragma unroll
        for (int j = 0; j < 8; ++j) { a[j] = (__bf16)x0[j]; a[j + 8] = (__bf16)x1[j]; }
#pragma unroll
        for (int u = 0; u < NTW; ++u) {
            const int nt = ng * NTW + u;
            const v16bf bm = *(const v16bf*)(wp +
                ((size_t)(nt * (CC / 32) + (kc >> 5)) * 32 + lane) * 16);
            acc[u] = __builtin_amdgcn_wmma_f32_16x16x32_bf16(false, a, false, bm,
                                                             (short)0, acc[u], false, false);
        }
    }

#pragma unroll
    for (int u = 0; u < NTW; ++u) {
        const int coln = (ng * NTW + u) * 16 + lq;
        const float bv = bias[coln];
        if (TRANS) {                         // hT: rows contiguous -> one 16B store
            v8bf st;
#pragma unroll
            for (int r = 0; r < 8; ++r) st[r] = (__bf16)(acc[u][r] + bv);
            *(v8bf*)(out + ((size_t)b * Co + coln) * NN + mt * 16 + lh * 8) = st;
        } else {                             // f,g: row-major, strided b16 stores
#pragma unroll
            for (int r = 0; r < 8; ++r)
                out[((size_t)b * NN + mt * 16 + lh * 8 + r) * Co + coln] =
                    (__bf16)(acc[u][r] + bv);
        }
    }
}

// ---------------------------------------------------------------------------
// Flash attention: block = one 16-query tile, 4 waves x 128 output channels.
// Scores computed TRANSPOSED (t[k,q] = f·gT) so the exp(t-m) values land
// exactly in the A-operand layout of the o += p@h WMMA (no shuffles/LDS).
// ---------------------------------------------------------------------------
__global__ __launch_bounds__(128)
void attn_kernel(const float* __restrict__ x, const __bf16* __restrict__ fmat,
                 const __bf16* __restrict__ gmat, const __bf16* __restrict__ hT,
                 const float* __restrict__ gamma, float* __restrict__ out) {
    const int lane = threadIdx.x & 31, wid = threadIdx.x >> 5;
    const int lq = lane & 15, lh = lane >> 4;
    const int b  = blockIdx.x >> 8;
    const int qt = blockIdx.x & 255;
    const int colbase = wid * 128;

    const __bf16* fb = fmat + (size_t)b * NN * CFG_;
    const __bf16* gb = gmat + (size_t)b * NN * CFG_;
    const __bf16* hb = hT   + (size_t)b * CC * NN;

    // gT B-operands (loop-invariant): elem j -> c = (j<8 ? lh*8+j : 16+lh*8+(j-8)) (+32)
    const __bf16* gr = gb + (size_t)(qt * 16 + lq) * CFG_;
    const v16bf bg0 = pack2(gr + lh * 8,      gr + 16 + lh * 8);
    const v16bf bg1 = pack2(gr + 32 + lh * 8, gr + 48 + lh * 8);

    v8f acc[8];
#pragma unroll
    for (int cg = 0; cg < 8; ++cg) acc[cg] = (v8f){};
    float m = -INFINITY, l = 0.0f;

    for (int k0 = 0; k0 < NN; k0 += 32) {
        const __bf16* fr0 = fb + (size_t)(k0 + lq) * CFG_;
        const __bf16* fr1 = fb + (size_t)(k0 + 16 + lq) * CFG_;
        v8f t0 = {}, t1 = {};
        t0 = __builtin_amdgcn_wmma_f32_16x16x32_bf16(false,
                 pack2(fr0 + lh * 8, fr0 + 16 + lh * 8), false, bg0, (short)0, t0, false, false);
        t0 = __builtin_amdgcn_wmma_f32_16x16x32_bf16(false,
                 pack2(fr0 + 32 + lh * 8, fr0 + 48 + lh * 8), false, bg1, (short)0, t0, false, false);
        t1 = __builtin_amdgcn_wmma_f32_16x16x32_bf16(false,
                 pack2(fr1 + lh * 8, fr1 + 16 + lh * 8), false, bg0, (short)0, t1, false, false);
        t1 = __builtin_amdgcn_wmma_f32_16x16x32_bf16(false,
                 pack2(fr1 + 32 + lh * 8, fr1 + 48 + lh * 8), false, bg1, (short)0, t1, false, false);

        // online softmax for query q = lq
        float pm = t0[0];
#pragma unroll
        for (int r = 1; r < 8; ++r) pm = fmaxf(pm, t0[r]);
#pragma unroll
        for (int r = 0; r < 8; ++r) pm = fmaxf(pm, t1[r]);
        pm = fmaxf(pm, __shfl_xor(pm, 16));
        const float mnew = fmaxf(m, pm);
        const float corr = __expf(m - mnew);
        m = mnew;

        float rs = 0.0f;
        v16bf ap;                            // p already in A-operand layout
#pragma unroll
        for (int r = 0; r < 8; ++r) {
            const float p0 = __expf(t0[r] - mnew);
            const float p1 = __expf(t1[r] - mnew);
            rs += p0 + p1;
            ap[r] = (__bf16)p0; ap[r + 8] = (__bf16)p1;
        }
        rs += __shfl_xor(rs, 16);
        l = l * corr + rs;

#pragma unroll
        for (int r = 0; r < 8; ++r) {
            const float cr = __shfl(corr, lh * 8 + r);
#pragma unroll
            for (int cg = 0; cg < 8; ++cg) acc[cg][r] *= cr;
        }

#pragma unroll
        for (int cg = 0; cg < 8; ++cg) {
            const __bf16* hr = hb + (size_t)(colbase + cg * 16 + lq) * NN + k0 + lh * 8;
            acc[cg] = __builtin_amdgcn_wmma_f32_16x16x32_bf16(false, ap, false,
                          pack2(hr, hr + 16), (short)0, acc[cg], false, false);
        }
    }

    // epilogue: out = gamma * (acc / l) + x
    const float gma = gamma[0];
    const float invl = 1.0f / l;
    float invr[8];
#pragma unroll
    for (int r = 0; r < 8; ++r) invr[r] = __shfl(invl, lh * 8 + r);
#pragma unroll
    for (int cg = 0; cg < 8; ++cg) {
#pragma unroll
        for (int r = 0; r < 8; ++r) {
            const int q = qt * 16 + lh * 8 + r;
            const int n = colbase + cg * 16 + lq;
            const size_t idx = ((size_t)b * NN + q) * CC + n;
            out[idx] = gma * acc[cg][r] * invr[r] + x[idx];
        }
    }
}

// ---------------------------------------------------------------------------
extern "C" void kernel_launch(void* const* d_in, const int* in_sizes, int n_in,
                              void* d_out, int out_size, void* d_ws, size_t ws_size,
                              hipStream_t stream) {
    const float* x     = (const float*)d_in[0];
    const float* kf    = (const float*)d_in[1];
    const float* kg    = (const float*)d_in[2];
    const float* kh    = (const float*)d_in[3];
    const float* bf_   = (const float*)d_in[4];
    const float* bg_   = (const float*)d_in[5];
    const float* bh_   = (const float*)d_in[6];
    const float* gamma = (const float*)d_in[7];
    float* out = (float*)d_out;

    // Workspace (bf16): f[B,N,64] | g[B,N,64] | hT[B,512,N] | packed weights
    __bf16* fws = (__bf16*)d_ws;
    __bf16* gws = fws + (size_t)BN * NN * CFG_;
    __bf16* hws = gws + (size_t)BN * NN * CFG_;
    __bf16* wpf = hws + (size_t)BN * CC * NN;
    __bf16* wpg = wpf + (size_t)CC * CFG_;
    __bf16* wph = wpg + (size_t)CC * CFG_;

    // Pack weights into B-operand bf16 layout (tiny).
    pack_w<<<(CC * CFG_) / 256, 256, 0, stream>>>(kf, wpf, CFG_);
    pack_w<<<(CC * CFG_) / 256, 256, 0, stream>>>(kg, wpg, CFG_);
    pack_w<<<(CC * CC)   / 256, 256, 0, stream>>>(kh, wph, CC);

    // f,g: 4 batches * 256 row-tiles * 1 col-group = 1024 waves -> 128 blocks
    proj_kernel<4, false><<<128, 256, 0, stream>>>(x, wpf, bf_, fws, CFG_);
    proj_kernel<4, false><<<128, 256, 0, stream>>>(x, wpg, bg_, gws, CFG_);
    // hT: 4 * 256 * 8 col-groups = 8192 waves -> 1024 blocks
    proj_kernel<4, true><<<1024, 256, 0, stream>>>(x, wph, bh_, hws, CC);
    // attention: one block per (batch, query tile)
    attn_kernel<<<BN * (NN / 16), 128, 0, stream>>>(x, fws, gws, hws, gamma, out);
}
